// HybridColorGenerator_60309930771205
// MI455X (gfx1250) — matrix operand order repaired and checked by reading, not compiled, not run on hardware
//
#include <hip/hip_runtime.h>
#include <cstdint>

// ============================================================================
// HybridColorGenerator for MI455X (gfx1250, wave32, WMMA).
//
// Pipeline (td branch is dead code in the reference -> skipped):
//   enc:   x(18) ->64 ->128 ->256        (Linear[no bias] + batch-BN + leaky)
//   prep:  light_in/dark_in = [f | x | base]  f16, K padded 328->352
//   refine(light/dark): 352->256 ->192 ->162(sigmoid)
//   rules: blend row0 with base, contrast fixups, write (B,324) f32
//
// BN trick: bias before batch-BN cancels; BN+leaky is folded into the NEXT
// GEMM's A-tile loader via precomputed (scale, shift). Column stats are
// reduced deterministically (per-block partials + fixed-order reduce kernel).
//
// GEMM: 128x64 tile / workgroup, 8 waves, each wave = 16 rows x 64 cols as
// four v_wmma_f32_16x16x32_f16 accumulators. A staged in LDS row-major
// (fragment = 2x ds_load_b128); W staged in LDS in fragment-ready layout
// (16 contiguous halves per lane per subtile -> 2x ds_load_b128).
// ============================================================================

typedef __attribute__((ext_vector_type(16))) _Float16 v16h;
typedef __attribute__((ext_vector_type(8)))  float    v8f;

#define THREADS   256
#define M_TILE    128
#define N_TILE    64
#define K_STEP    32
#define AS_STRIDE 40   // halves; padded row stride to spread LDS banks

union AFrag { v16h v; unsigned int u[8]; uint4 q[2]; };

__device__ __forceinline__ float leaky02(float v) { return v > 0.f ? v : 0.2f * v; }
__device__ __forceinline__ float clamp01(float v) { return fminf(fmaxf(v, 0.f), 1.f); }

// ---------------------------------------------------------------------------
// Fused tile GEMM.
// MODE 0: A = raw f32, Kreal may be < Kpad (zero-fill); scalar guarded loads
// MODE 1: A = raw f32 pre-BN activations, Kreal == Kpad (vector loads);
//         apply leaky(a*scale+shift) on load
// MODE 2: A = pre-materialized f16 (row stride = Kpad), vector loads
// FINAL false: store raw f32 C + deterministic per-block col sum/sumsq
// FINAL true : store sigmoid(C + bias)
// ---------------------------------------------------------------------------
template<int MODE, bool FINAL>
__global__ __launch_bounds__(THREADS)
void gemm_tile(const void* __restrict__ Ain,
               const _Float16* __restrict__ Wh,       // Kpad x Npad, zero-padded
               const float* __restrict__ inScale,
               const float* __restrict__ inShift,
               const float* __restrict__ bias, int Nreal,
               float* __restrict__ Cout,              // B x Npad
               float* __restrict__ pSum, float* __restrict__ pSq, // gridM x 256
               int Kreal, int Kpad, int Npad)
{
  __shared__ _Float16 As[M_TILE * AS_STRIDE];
  // W tile, fragment-ready: [subtile t][lane][16 halves] -> 32B/lane contiguous
  __shared__ _Float16 Bs[4 * 32 * 16];
  __shared__ float redS[8][N_TILE];
  __shared__ float redQ[8][N_TILE];

  const int lane  = threadIdx.x & 31;
  const int wid   = threadIdx.x >> 5;
  const int mBase = blockIdx.x * M_TILE;
  const int nBase = blockIdx.y * N_TILE;

  v8f acc[4] = {};

  const int ldRow = threadIdx.x >> 1;        // 0..127 : one row per 2 threads
  const int ldKg  = (threadIdx.x & 1) << 4;  // 0 or 16 (half-index group)

  for (int kk = 0; kk < Kpad; kk += K_STEP) {
    // ---- stage A tile (128 rows x 32 halves) into LDS ----
    if (MODE == 2) {
      const _Float16* A16 = (const _Float16*)Ain
                          + (size_t)(mBase + ldRow) * Kpad + kk + ldKg;
      const uint4* s = (const uint4*)A16;          // 32B aligned (stride 352h)
      uint4 p0 = s[0], p1 = s[1];
      uint4* d = (uint4*)&As[ldRow * AS_STRIDE + ldKg];
      d[0] = p0; d[1] = p1;
    } else if (MODE == 1) {
      // Kreal == Kpad (64/128/192/256): unguarded 4x float4 loads, BN fold.
      const float* A32 = (const float*)Ain + (size_t)(mBase + ldRow) * Kreal
                       + kk + ldKg;
      float4 f[4];
      #pragma unroll
      for (int p = 0; p < 4; ++p) f[p] = ((const float4*)A32)[p];
      const float* fp = (const float*)f;
      #pragma unroll
      for (int j = 0; j < 16; ++j) {
        int gk = kk + ldKg + j;
        As[ldRow * AS_STRIDE + ldKg + j] =
            (_Float16)leaky02(fp[j] * inScale[gk] + inShift[gk]);
      }
    } else {
      const float* A32 = (const float*)Ain + (size_t)(mBase + ldRow) * Kreal;
      #pragma unroll
      for (int j = 0; j < 16; ++j) {
        int gk = kk + ldKg + j;
        float v = (gk < Kreal) ? A32[gk] : 0.f;
        As[ldRow * AS_STRIDE + ldKg + j] = (_Float16)v;
      }
    }
    // ---- stage B tile (K_STEP x N_TILE) in fragment-ready layout ----
    // element (k, n) lives at lane = (n&15) + ((k&8)<<1), half = (k&7) + ((k&16)>>1),
    // subtile t = n>>4  ==> Bs[((n>>4)*32 + ((k&8)<<1) + (n&15))*16 + (k&7) + ((k&16)>>1)]
    for (int idx = threadIdx.x; idx < N_TILE * K_STEP; idx += THREADS) {
      int n = idx >> 5, k = idx & 31;
      Bs[(((n >> 4) << 5) + ((k & 8) << 1) + (n & 15)) * 16 + (k & 7) + ((k & 16) >> 1)] =
          Wh[(size_t)(kk + k) * Npad + nBase + n];
    }
    __syncthreads();

    // ---- A fragment (16-bit A 16x32 layout, ISA 7.12.2) ----
    AFrag a;
    {
      int row  = (wid << 4) + (lane & 15);
      int koff = (lane & 16) ? 8 : 0;
      const _Float16* ap = &As[row * AS_STRIDE + koff];
      a.q[0] = *(const uint4*)(ap);        // halves koff+0..7   (K 0..7 / 8..15)
      a.q[1] = *(const uint4*)(ap + 16);   // halves koff+16..23 (K 16..23/24..31)
    }
    // ---- 4 x (B fragment + WMMA) covering 64 columns ----
    #pragma unroll
    for (int t = 0; t < 4; ++t) {
      AFrag b;
      const _Float16* bp = &Bs[((t << 5) + lane) << 4];  // 32B contiguous
      b.q[0] = *(const uint4*)(bp);
      b.q[1] = *(const uint4*)(bp + 8);
      acc[t] = __builtin_amdgcn_wmma_f32_16x16x32_f16(
                   false, a.v, false, b.v, (short)0, acc[t], false, false);
    }
    __syncthreads();
  }

  // ---- epilogue: store (+ stats or sigmoid) ----
  const int rowB = mBase + (wid << 4) + ((lane & 16) ? 8 : 0);
  #pragma unroll
  for (int t = 0; t < 4; ++t) {
    int gcol = nBase + (t << 4) + (lane & 15);
    float s = 0.f, q = 0.f;
    #pragma unroll
    for (int r = 0; r < 8; ++r) {
      float v = acc[t][r];
      if (FINAL) {
        float bb = (gcol < Nreal) ? bias[gcol] : 0.f;
        v += bb;
        v = 1.f / (1.f + __expf(-v));
      } else {
        s += v; q += v * v;
      }
      Cout[(size_t)(rowB + r) * Npad + gcol] = v;
    }
    if (!FINAL) {
      s += __shfl_xor(s, 16);   // combine the two 8-row halves of this column
      q += __shfl_xor(q, 16);
      if (lane < 16) { redS[wid][(t << 4) + lane] = s; redQ[wid][(t << 4) + lane] = q; }
    }
  }
  if (!FINAL) {
    __syncthreads();
    if (threadIdx.x < N_TILE) {
      float s = 0.f, q = 0.f;
      #pragma unroll
      for (int w = 0; w < 8; ++w) { s += redS[w][threadIdx.x]; q += redQ[w][threadIdx.x]; }
      int gcol = nBase + threadIdx.x;
      pSum[blockIdx.x * 256 + gcol] = s;   // deterministic per-block partials
      pSq [blockIdx.x * 256 + gcol] = q;
    }
  }
}

// Fixed-order reduction of per-block partials -> folded BN (scale, shift).
__global__ void bn_stats(const float* __restrict__ pSum, const float* __restrict__ pSq,
                         const float* __restrict__ g, const float* __restrict__ b,
                         float* __restrict__ scale, float* __restrict__ shift,
                         int blocks, float invB)
{
  int c = threadIdx.x;
  float s = 0.f, q = 0.f;
  for (int i = 0; i < blocks; ++i) { s += pSum[i * 256 + c]; q += pSq[i * 256 + c]; }
  float mean = s * invB;
  float var  = q * invB - mean * mean;
  float r    = rsqrtf(var + 1e-5f);
  float sc   = g[c] * r;
  scale[c] = sc;
  shift[c] = b[c] - mean * sc;
}

// f32 weight (Kreal x Nreal, row-major) -> zero-padded f16 (Kpad x Npad).
__global__ void convert_w(const float* __restrict__ W, _Float16* __restrict__ Wh,
                          int Kreal, int Nreal, int Kpad, int Npad)
{
  int i = blockIdx.x * blockDim.x + threadIdx.x;
  if (i >= Kpad * Npad) return;
  int k = i / Npad, n = i - k * Npad;
  float v = (k < Kreal && n < Nreal) ? W[(size_t)k * Nreal + n] : 0.f;
  Wh[i] = (_Float16)v;
}

// Materialize refine input [BN(h3)+leaky | x | base | 0-pad] as f16, K=352.
__global__ void prep_in(const float* __restrict__ x, const float* __restrict__ h3,
                        const float* __restrict__ sc, const float* __restrict__ sh,
                        const float* __restrict__ pTint, const float* __restrict__ pDbg,
                        const float* __restrict__ pLtext,
                        _Float16* __restrict__ out16, int isDark, int total)
{
  int i = blockIdx.x * blockDim.x + threadIdx.x;
  if (i >= total) return;
  int row = i / 352;
  int c   = i - row * 352;
  float v = 0.f;
  if (c < 256) {
    float h = h3[(size_t)row * 256 + c];
    v = leaky02(h * sc[c] + sh[c]);
  } else if (c < 274) {
    v = x[(size_t)row * 18 + (c - 256)];
  } else if (c < 328) {
    int j   = c - 274;              // flat (6,3,3)
    int cg  = j / 9;
    int rem = j - cg * 9;
    int rr  = rem / 3;
    int k   = rem - rr * 3;
    if (rr == 0) {
      float xv = x[(size_t)row * 18 + cg * 3 + k];
      if (!isDark) {
        v = xv;
      } else {
        float dbg = fminf(fmaxf(pDbg[0], 0.05f), 0.25f);
        float lt  = fminf(fmaxf(pLtext[0], 0.75f), 0.95f);
        float ti  = fminf(fmaxf(pTint[0], 0.01f), 0.2f);
        if (cg == 0 || cg == 3) {
          float xa = x[(size_t)row * 18 + (cg + 2) * 3 + k];
          v = clamp01(dbg + 0.05f * (1.f - xv) + ti * xa);
        } else if (cg == 1 || cg == 4) {
          v = lt + 0.15f * (1.f - xv);
        } else {
          v = clamp01(xv * 1.3f);
        }
      }
    }
  }
  out16[(size_t)row * 352 + c] = (_Float16)v;
}

// Copy-through + _apply_rules for both themes; final (B,324) output.
__global__ __launch_bounds__(64)
void epilogue(const float* __restrict__ lout, const float* __restrict__ dout,
              const float* __restrict__ x,
              const float* __restrict__ pTint, const float* __restrict__ pDbg,
              const float* __restrict__ pLtext,
              float* __restrict__ out)
{
  const int row = blockIdx.x;
  float* orow = out + (size_t)row * 324;
  for (int e = threadIdx.x; e < 162; e += 64) {
    orow[e]       = lout[(size_t)row * 192 + e];
    orow[162 + e] = dout[(size_t)row * 192 + e];
  }
  __syncthreads();
  if (threadIdx.x < 2) {
    const bool isDark = (threadIdx.x == 1);
    float* o = orow + (isDark ? 162 : 0);
    const float* xr = x + (size_t)row * 18;
    float dbg = fminf(fmaxf(pDbg[0], 0.05f), 0.25f);
    float lt  = fminf(fmaxf(pLtext[0], 0.75f), 0.95f);
    float ti  = fminf(fmaxf(pTint[0], 0.01f), 0.2f);
    float r0[6][3];
    for (int cg = 0; cg < 6; ++cg)
      for (int k = 0; k < 3; ++k) {
        float xv = xr[cg * 3 + k];
        float base;
        if (!isDark)                 base = xv;
        else if (cg == 0 || cg == 3) base = clamp01(dbg + 0.05f * (1.f - xv) + ti * xr[(cg + 2) * 3 + k]);
        else if (cg == 1 || cg == 4) base = lt + 0.15f * (1.f - xv);
        else                         base = clamp01(xv * 1.3f);
        r0[cg][k] = 0.7f * base + 0.3f * o[cg * 27 + k];
      }
    const float mc = isDark ? 0.6f : 0.5f;
    for (int ii = 0; ii < 2; ++ii) {
      int i = (ii == 0) ? 1 : 4;
      float bgl = 0.299f * r0[i-1][0] + 0.587f * r0[i-1][1] + 0.114f * r0[i-1][2];
      float tl  = 0.299f * r0[i][0]   + 0.587f * r0[i][1]   + 0.114f * r0[i][2];
      bool need   = fabsf(bgl - tl) < mc;
      bool branch = (bgl < 0.5f) != isDark;
      if (need)
        for (int k = 0; k < 3; ++k)
          r0[i][k] = branch ? clamp01(r0[i][k] * 0.7f) : clamp01(r0[i][k] + 0.3f);
    }
    for (int cg = 0; cg < 6; ++cg)
      for (int k = 0; k < 3; ++k)
        o[cg * 27 + k] = r0[cg][k];
  }
}

// ============================================================================
extern "C" void kernel_launch(void* const* d_in, const int* in_sizes, int n_in,
                              void* d_out, int out_size, void* d_ws, size_t ws_size,
                              hipStream_t stream)
{
  const int B     = in_sizes[0] / 18;     // 131072
  const int gridM = B / M_TILE;           // 1024
  const float invB = 1.f / (float)B;

  // d_in layout (setup_inputs order): 0=x, td:1..10 (dead), enc:11..22,
  // light:23..32, dark:33..42, 43=tint, 44=dark_bg, 45=light_text.
  const float* x     = (const float*)d_in[0];
  const float* tint  = (const float*)d_in[43];
  const float* dbg   = (const float*)d_in[44];
  const float* ltext = (const float*)d_in[45];

  // ---- workspace carve-up (~760 MB; small stuff first) ----
  char* ws = (char*)d_ws;
  size_t cur = 0;
  auto alloc = [&](size_t bytes) -> char* {
    char* p = ws + cur;
    cur += (bytes + 255) & ~(size_t)255;
    return p;
  };
  float* scaleA = (float*)alloc(256 * 4);
  float* shiftA = (float*)alloc(256 * 4);
  float* scale3 = (float*)alloc(256 * 4);
  float* shift3 = (float*)alloc(256 * 4);
  float* pSum   = (float*)alloc((size_t)gridM * 256 * 4);
  float* pSq    = (float*)alloc((size_t)gridM * 256 * 4);
  _Float16* whE0 = (_Float16*)alloc(32  * 64  * 2);
  _Float16* whE1 = (_Float16*)alloc(64  * 128 * 2);
  _Float16* whE2 = (_Float16*)alloc(128 * 256 * 2);
  _Float16* whL0 = (_Float16*)alloc(352 * 256 * 2);
  _Float16* whL1 = (_Float16*)alloc(256 * 192 * 2);
  _Float16* whL2 = (_Float16*)alloc(192 * 192 * 2);
  _Float16* whD0 = (_Float16*)alloc(352 * 256 * 2);
  _Float16* whD1 = (_Float16*)alloc(256 * 192 * 2);
  _Float16* whD2 = (_Float16*)alloc(192 * 192 * 2);
  float* h1   = (float*)alloc((size_t)B * 64  * 4);
  float* h2   = (float*)alloc((size_t)B * 128 * 4);
  float* h3   = (float*)alloc((size_t)B * 256 * 4);
  _Float16* in16 = (_Float16*)alloc((size_t)B * 352 * 2);
  float* t1   = (float*)alloc((size_t)B * 256 * 4);
  float* t2   = (float*)alloc((size_t)B * 192 * 4);
  float* loutB = (float*)alloc((size_t)B * 192 * 4);
  float* doutB = (float*)alloc((size_t)B * 192 * 4);

  auto cvt = [&](int idx, _Float16* Wh, int Kr, int Nr, int Kp, int Np) {
    int tot = Kp * Np;
    convert_w<<<(tot + 255) / 256, 256, 0, stream>>>((const float*)d_in[idx], Wh, Kr, Nr, Kp, Np);
  };
  cvt(11, whE0, 18,  64,  32,  64);
  cvt(15, whE1, 64,  128, 64,  128);
  cvt(19, whE2, 128, 256, 128, 256);
  cvt(23, whL0, 328, 256, 352, 256);
  cvt(27, whL1, 256, 192, 256, 192);
  cvt(31, whL2, 192, 162, 192, 192);
  cvt(33, whD0, 328, 256, 352, 256);
  cvt(37, whD1, 256, 192, 256, 192);
  cvt(41, whD2, 192, 162, 192, 192);

  dim3 blk(THREADS);
  auto stats = [&](int N, int gi, int bi, float* sc, float* sh) {
    bn_stats<<<1, N, 0, stream>>>(pSum, pSq, (const float*)d_in[gi],
                                  (const float*)d_in[bi], sc, sh, gridM, invB);
  };

  // ---- encoder ----
  gemm_tile<0, false><<<dim3(gridM, 1), blk, 0, stream>>>(
      x, whE0, nullptr, nullptr, nullptr, 64, h1, pSum, pSq, 18, 32, 64);
  stats(64, 13, 14, scaleA, shiftA);
  gemm_tile<1, false><<<dim3(gridM, 2), blk, 0, stream>>>(
      h1, whE1, scaleA, shiftA, nullptr, 128, h2, pSum, pSq, 64, 64, 128);
  stats(128, 17, 18, scaleA, shiftA);
  gemm_tile<1, false><<<dim3(gridM, 4), blk, 0, stream>>>(
      h2, whE2, scaleA, shiftA, nullptr, 256, h3, pSum, pSq, 128, 128, 256);
  stats(256, 21, 22, scale3, shift3);   // kept live for both prep passes

  const int prepTot  = B * 352;
  const int prepGrid = (prepTot + 255) / 256;

  // ---- light branch ----
  prep_in<<<prepGrid, 256, 0, stream>>>(x, h3, scale3, shift3, tint, dbg, ltext, in16, 0, prepTot);
  gemm_tile<2, false><<<dim3(gridM, 4), blk, 0, stream>>>(
      in16, whL0, nullptr, nullptr, nullptr, 256, t1, pSum, pSq, 352, 352, 256);
  stats(256, 25, 26, scaleA, shiftA);
  gemm_tile<1, false><<<dim3(gridM, 3), blk, 0, stream>>>(
      t1, whL1, scaleA, shiftA, nullptr, 192, t2, pSum, pSq, 256, 256, 192);
  stats(192, 29, 30, scaleA, shiftA);
  gemm_tile<1, true><<<dim3(gridM, 3), blk, 0, stream>>>(
      t2, whL2, scaleA, shiftA, (const float*)d_in[32], 162, loutB, pSum, pSq, 192, 192, 192);

  // ---- dark branch (reuses in16/t1/t2) ----
  prep_in<<<prepGrid, 256, 0, stream>>>(x, h3, scale3, shift3, tint, dbg, ltext, in16, 1, prepTot);
  gemm_tile<2, false><<<dim3(gridM, 4), blk, 0, stream>>>(
      in16, whD0, nullptr, nullptr, nullptr, 256, t1, pSum, pSq, 352, 352, 256);
  stats(256, 35, 36, scaleA, shiftA);
  gemm_tile<1, false><<<dim3(gridM, 3), blk, 0, stream>>>(
      t1, whD1, scaleA, shiftA, nullptr, 192, t2, pSum, pSq, 256, 256, 192);
  stats(192, 39, 40, scaleA, shiftA);
  gemm_tile<1, true><<<dim3(gridM, 3), blk, 0, stream>>>(
      t2, whD2, scaleA, shiftA, (const float*)d_in[42], 162, doutB, pSum, pSq, 192, 192, 192);

  // ---- rules + final output ----
  epilogue<<<B, 64, 0, stream>>>(loutB, doutB, x, tint, dbg, ltext, (float*)d_out);
}
